// GhostVlad_77790447665615
// MI455X (gfx1250) — matrix operand order repaired and checked by reading, not compiled, not run on hardware
//
#include <hip/hip_runtime.h>
#include <hip/hip_bf16.h>

typedef __bf16 bf16_t;
typedef __attribute__((ext_vector_type(4)))  __bf16 v4bf;
typedef __attribute__((ext_vector_type(16))) __bf16 v16bf;
typedef __attribute__((ext_vector_type(8)))  float  v8f;

#define N_   32
#define D_   512
#define T_   2048
#define K_   64
#define KG_  66
#define MPAD 80     // 66 padded to 5 WMMA M-tiles
#define TCH  128    // t-chunk per block in kernel 1
#define EPS_ 1e-12f

// ---- WMMA fragment loaders (wave32 layouts, CDNA5 ISA 7.12.2) ----
// A: 16x32 bf16 (MxK). lane&15 = row M; lanes<16 hold K 0-7,16-23; lanes>=16 hold K 8-15,24-31.
// Per-lane reads are two contiguous 16B runs -> ds_load_b128 pairs.
__device__ __forceinline__ v16bf frag_A(const bf16_t* t, int stride, int lane) {
    v16bf a;
    const int m  = lane & 15;
    const int ko = (lane < 16) ? 0 : 8;
#pragma unroll
    for (int e = 0; e < 16; ++e) {
        int k = e + ((e < 8) ? 0 : 8) + ko;
        a[e] = t[m * stride + k];
    }
    return a;
}
// B: 32x16 bf16 (KxN). lane&15 = N column; lanes<16 hold K 0-7,16-23; lanes>=16 hold K 8-15,24-31.
__device__ __forceinline__ v16bf frag_B(const bf16_t* t, int stride, int lane) {
    v16bf b;
    const int nn = lane & 15;
    const int ko = (lane < 16) ? 0 : 8;
#pragma unroll
    for (int e = 0; e < 16; ++e) {
        int k = e + ((e < 8) ? 0 : 8) + ko;
        b[e] = t[k * stride + nn];
    }
    return b;
}

// =====================================================================
// Kernel 1: per (n, t-chunk): L2 norm over D, logits = conv_w @ xn via
// bf16 WMMA, softmax over 66 clusters, store soft_assign (bf16) + 1/norm.
// =====================================================================
__global__ __launch_bounds__(256) void k_assign(
    const float* __restrict__ x, const float* __restrict__ cw,
    const float* __restrict__ cb, float* __restrict__ invn,
    bf16_t* __restrict__ assign)
{
    __shared__ float  s_red[1024];        // 4 partial sums per thread (norm pass)
    __shared__ float  s_inv[TCH];
    __shared__ bf16_t s_x[32 * TCH];      // xn K-slab, [32][TCH]
    __shared__ bf16_t s_w[MPAD * 32];     // conv_w K-slab, [80][32]
    __shared__ float  s_l[MPAD * TCH];    // logits, [80][TCH] (40 KB)
    __shared__ float  s_m[TCH];
    __shared__ float  s_d[TCH];

    const int tid  = threadIdx.x;
    const int lane = tid & 31;
    const int wave = tid >> 5;
    const int n    = blockIdx.y;
    const int t0   = blockIdx.x * TCH;

    // ---- channel L2 norm: float4 along t, 8 d-partitions per t-quad ----
    {
        const int q    = tid & 31;   // t-quad (coalesced across the wave)
        const int part = tid >> 5;   // 64-deep d partition
        const float* xp = x + (size_t)(n * D_ + part * 64) * T_ + t0 + q * 4;
        float s0 = 0.f, s1 = 0.f, s2 = 0.f, s3 = 0.f;
#pragma unroll 4
        for (int d = 0; d < 64; ++d) {
            float4 v = *(const float4*)(xp + (size_t)d * T_);
            s0 += v.x * v.x; s1 += v.y * v.y; s2 += v.z * v.z; s3 += v.w * v.w;
        }
        s_red[tid * 4 + 0] = s0; s_red[tid * 4 + 1] = s1;
        s_red[tid * 4 + 2] = s2; s_red[tid * 4 + 3] = s3;
        // zero-fill ghost rows 66..79 of s_w once (448 bf16 = 112 x v4bf)
        if (tid < 112) {
            v4bf z = {(bf16_t)0.f, (bf16_t)0.f, (bf16_t)0.f, (bf16_t)0.f};
            *(v4bf*)&s_w[KG_ * 32 + tid * 4] = z;
        }
        __syncthreads();
        if (tid < TCH) {
            float tot = 0.f;
#pragma unroll
            for (int p = 0; p < 8; ++p) tot += s_red[(p * 32 + (tid >> 2)) * 4 + (tid & 3)];
            float inv = 1.f / fmaxf(sqrtf(tot), EPS_);
            s_inv[tid] = inv;
            invn[n * T_ + t0 + tid] = inv;
        }
        __syncthreads();
    }

    // ---- logits GEMM: [80x512] x [512xTCH]; K-slabs of 32, reg-prefetch ----
    v8f acc[5];
#pragma unroll
    for (int i = 0; i < 5; ++i) acc[i] = (v8f){0.f,0.f,0.f,0.f,0.f,0.f,0.f,0.f};

    float4 xr[4];                        // prefetched xn slab (16 floats/thread)
#pragma unroll
    for (int j = 0; j < 4; ++j) {
        int idx = (j * 256 + tid) * 4;
        xr[j] = *(const float4*)&x[(size_t)(n * D_ + (idx >> 7)) * T_ + t0 + (idx & 127)];
    }

    for (int kk = 0; kk < D_; kk += 32) {
        // stage conv_w slab rows < 66 (no predication; float4 loads)
        for (int i = tid; i < KG_ * 8; i += 256) {
            int g = i >> 3, qq = (i & 7) * 4;
            float4 w = *(const float4*)&cw[g * D_ + kk + qq];
            v4bf wb; wb[0] = (bf16_t)w.x; wb[1] = (bf16_t)w.y;
                     wb[2] = (bf16_t)w.z; wb[3] = (bf16_t)w.w;
            *(v4bf*)&s_w[g * 32 + qq] = wb;
        }
        // commit prefetched xn slab to LDS (scaled, packed bf16 x4)
#pragma unroll
        for (int j = 0; j < 4; ++j) {
            int idx = (j * 256 + tid) * 4;
            int t   = idx & 127;
            float4 iv = *(const float4*)&s_inv[t];
            v4bf xb; xb[0] = (bf16_t)(xr[j].x * iv.x); xb[1] = (bf16_t)(xr[j].y * iv.y);
                     xb[2] = (bf16_t)(xr[j].z * iv.z); xb[3] = (bf16_t)(xr[j].w * iv.w);
            *(v4bf*)&s_x[idx] = xb;
        }
        __syncthreads();
        if (kk + 32 < D_) {              // prefetch next slab while WMMA runs
#pragma unroll
            for (int j = 0; j < 4; ++j) {
                int idx = (j * 256 + tid) * 4;
                xr[j] = *(const float4*)&x[(size_t)(n * D_ + kk + 32 + (idx >> 7)) * T_ + t0 + (idx & 127)];
            }
        }
        v16bf bf = frag_B(s_x + wave * 16, TCH, lane);   // wave owns 16 t-columns
#pragma unroll
        for (int mt = 0; mt < 5; ++mt) {
            v16bf af = frag_A(s_w + mt * 16 * 32, 32, lane);
            acc[mt] = __builtin_amdgcn_wmma_f32_16x16x32_bf16(
                false, af, false, bf, (short)0, acc[mt], false, false);
        }
        __syncthreads();
    }

    // spill logits (+bias) to LDS
    {
        const int col = wave * 16 + (lane & 15);
        const int mro = (lane < 16) ? 0 : 8;
#pragma unroll
        for (int mt = 0; mt < 5; ++mt)
#pragma unroll
            for (int r = 0; r < 8; ++r) {
                int g = mt * 16 + mro + r;
                s_l[g * TCH + col] = acc[mt][r] + ((g < KG_) ? cb[g] : 0.f);
            }
    }
    __syncthreads();

    // softmax over 66 clusters per column (ghosts in denominator only)
    if (tid < TCH) {
        float m = -3.4e38f;
        for (int g = 0; g < KG_; ++g) m = fmaxf(m, s_l[g * TCH + tid]);
        float den = 0.f;
        for (int g = 0; g < KG_; ++g) den += __expf(s_l[g * TCH + tid] - m);
        s_m[tid] = m;
        s_d[tid] = 1.f / den;
    }
    __syncthreads();
    // drop ghosts; packed bf16 x4 stores
    for (int i = tid; i < K_ * TCH / 4; i += 256) {
        int idx = i * 4, k = idx >> 7, t = idx & 127;
        float4 mm = *(const float4*)&s_m[t];
        float4 dd = *(const float4*)&s_d[t];
        v4bf ab;
        ab[0] = (bf16_t)(__expf(s_l[k * TCH + t + 0] - mm.x) * dd.x);
        ab[1] = (bf16_t)(__expf(s_l[k * TCH + t + 1] - mm.y) * dd.y);
        ab[2] = (bf16_t)(__expf(s_l[k * TCH + t + 2] - mm.z) * dd.z);
        ab[3] = (bf16_t)(__expf(s_l[k * TCH + t + 3] - mm.w) * dd.w);
        *(v4bf*)&assign[(size_t)(n * K_ + k) * T_ + t0 + t] = ab;
    }
}

// =====================================================================
// Kernel 2: agg[n,k,d] = sum_t a[n,k,t] * xn[n,d,t] via bf16 WMMA.
// One block per (n, 128-wide d tile); T-slabs of 32, reg-prefetch.
// =====================================================================
__global__ __launch_bounds__(256) void k_agg(
    const float* __restrict__ x, const float* __restrict__ invn,
    const bf16_t* __restrict__ assign, float* __restrict__ agg)
{
    __shared__ bf16_t s_a[K_ * 32];     // [64][32]  (k x t)
    __shared__ bf16_t s_b[32 * 128];    // [32][128] (t x d)
    __shared__ float  s_it[T_];         // per-n inverse norms (8 KB)

    const int tid  = threadIdx.x;
    const int lane = tid & 31;
    const int wave = tid >> 5;
    const int n    = blockIdx.y;
    const int d0   = blockIdx.x * 128;

    for (int i = tid; i < T_; i += 256) s_it[i] = invn[n * T_ + i];

    v8f acc[4];
#pragma unroll
    for (int i = 0; i < 4; ++i) acc[i] = (v8f){0.f,0.f,0.f,0.f,0.f,0.f,0.f,0.f};

    v4bf   ar[2];   // soft_assign slab (8 bf16/thread)
    float4 br[4];   // x slab (16 floats/thread)
    auto prefetch = [&](int t) {
#pragma unroll
        for (int j = 0; j < 2; ++j) {
            int idx = (j * 256 + tid) * 4;
            ar[j] = *(const v4bf*)&assign[(size_t)(n * K_ + (idx >> 5)) * T_ + t + (idx & 31)];
        }
#pragma unroll
        for (int j = 0; j < 4; ++j) {
            int idx = (j * 256 + tid) * 4;
            br[j] = *(const float4*)&x[(size_t)(n * D_ + d0 + (idx >> 5)) * T_ + t + (idx & 31)];
        }
    };

    prefetch(0);
    __syncthreads();                    // covers s_it
    for (int t = 0; t < T_; t += 32) {
        // commit prefetched slab to LDS
#pragma unroll
        for (int j = 0; j < 2; ++j) {
            int idx = (j * 256 + tid) * 4;
            *(v4bf*)&s_a[idx] = ar[j];
        }
#pragma unroll
        for (int j = 0; j < 4; ++j) {
            int idx = (j * 256 + tid) * 4;
            int dc = idx >> 5, ti = idx & 31;
            float4 iv = *(const float4*)&s_it[t + ti];
            s_b[(ti + 0) * 128 + dc] = (bf16_t)(br[j].x * iv.x);
            s_b[(ti + 1) * 128 + dc] = (bf16_t)(br[j].y * iv.y);
            s_b[(ti + 2) * 128 + dc] = (bf16_t)(br[j].z * iv.z);
            s_b[(ti + 3) * 128 + dc] = (bf16_t)(br[j].w * iv.w);
        }
        __syncthreads();
        if (t + 32 < T_) prefetch(t + 32);   // overlap with WMMA
        v16bf bf = frag_B(s_b + wave * 16, 128, lane);   // wave owns 16 d-columns
#pragma unroll
        for (int mt = 0; mt < 4; ++mt) {
            v16bf af = frag_A(s_a + mt * 16 * 32, 32, lane);
            acc[mt] = __builtin_amdgcn_wmma_f32_16x16x32_bf16(
                false, af, false, bf, (short)0, acc[mt], false, false);
        }
        __syncthreads();
    }

    const int col = d0 + wave * 16 + (lane & 15);
    const int mro = (lane < 16) ? 0 : 8;
#pragma unroll
    for (int mt = 0; mt < 4; ++mt)
#pragma unroll
        for (int r = 0; r < 8; ++r)
            agg[(size_t)(n * K_ + mt * 16 + mro + r) * D_ + col] = acc[mt][r];
}

// =====================================================================
// Kernel 3: a_sum, residual vs centroid, per-(n,k) L2 normalize over D.
// One block per (n,k).
// =====================================================================
__global__ __launch_bounds__(256) void k_final(
    const bf16_t* __restrict__ assign, const float* __restrict__ agg,
    const float* __restrict__ cent, float* __restrict__ out)
{
    __shared__ float s_red[256];
    __shared__ float s_bc[2];
    const int tid = threadIdx.x;
    const int k = blockIdx.x, n = blockIdx.y;
    const size_t row = (size_t)(n * K_ + k);

    float ps = 0.f;
    for (int t = tid * 4; t < T_; t += 1024) {
        v4bf a4 = *(const v4bf*)&assign[row * T_ + t];
        ps += (float)a4[0] + (float)a4[1] + (float)a4[2] + (float)a4[3];
    }
    s_red[tid] = ps;
    __syncthreads();
    for (int off = 128; off > 0; off >>= 1) {
        if (tid < off) s_red[tid] += s_red[tid + off];
        __syncthreads();
    }
    if (tid == 0) s_bc[0] = s_red[0];
    __syncthreads();
    const float asum = s_bc[0];
    __syncthreads();

    float v0 = agg[row * D_ + tid]       - asum * cent[k * D_ + tid];
    float v1 = agg[row * D_ + tid + 256] - asum * cent[k * D_ + tid + 256];
    s_red[tid] = v0 * v0 + v1 * v1;
    __syncthreads();
    for (int off = 128; off > 0; off >>= 1) {
        if (tid < off) s_red[tid] += s_red[tid + off];
        __syncthreads();
    }
    if (tid == 0) s_bc[1] = 1.f / fmaxf(sqrtf(s_red[0]), EPS_);
    __syncthreads();
    const float inv = s_bc[1];
    out[row * D_ + tid]       = v0 * inv;
    out[row * D_ + tid + 256] = v1 * inv;
}

// =====================================================================
extern "C" void kernel_launch(void* const* d_in, const int* in_sizes, int n_in,
                              void* d_out, int out_size, void* d_ws, size_t ws_size,
                              hipStream_t stream) {
    (void)in_sizes; (void)n_in; (void)out_size; (void)ws_size;
    const float* x    = (const float*)d_in[0];   // [32,512,2048]
    const float* cw   = (const float*)d_in[1];   // [66,512]
    const float* cb   = (const float*)d_in[2];   // [66]
    const float* cent = (const float*)d_in[3];   // [64,512]
    float* out = (float*)d_out;                  // [32, 64*512]

    // workspace: invn [N,T] f32 (256 KB) | assign [N,K,T] bf16 (8 MB) | agg [N,K,D] f32 (4 MB)
    char* ws = (char*)d_ws;
    float*  invn   = (float*)ws;
    bf16_t* assign = (bf16_t*)(ws + 256u * 1024u);
    float*  agg    = (float*)(ws + 256u * 1024u + 8u * 1024u * 1024u);

    k_assign<<<dim3(T_ / TCH, N_), 256, 0, stream>>>(x, cw, cb, invn, assign);
    k_agg   <<<dim3(D_ / 128, N_), 256, 0, stream>>>(x, invn, assign, agg);
    k_final <<<dim3(K_, N_),       256, 0, stream>>>(assign, agg, cent, out);
}
